// Qwen2MoeSparseMoeBlock_16587163697446
// MI455X (gfx1250) — matrix operand order repaired and checked by reading, not compiled, not run on hardware
//
#include <hip/hip_runtime.h>

// ---------------- problem constants ----------------
#define H_DIM   2048
#define E_NUM   16
#define TOPK    4
#define FE_DIM  1408
#define FS_DIM  5632
#define T_TOK   4096
#define SLOTS   (T_TOK * TOPK)   // total (token,expert) assignments = 16384

typedef __bf16 bf16_t;
typedef __bf16 v16bf __attribute__((ext_vector_type(16)));
typedef float  v8f   __attribute__((ext_vector_type(8)));
typedef unsigned int v4u  __attribute__((ext_vector_type(4)));
typedef int          v8i  __attribute__((ext_vector_type(8)));
typedef int          v4i  __attribute__((ext_vector_type(4)));

union FragAB { uint4 q[2]; v16bf v; };      // 32 B: one WMMA A/B fragment per lane
union FragC  { float f[8];  v8f  v; };      // 32 B: one WMMA C/D fragment per lane
union Pack8  { bf16_t h[8]; uint4 q; };     // 8 bf16 <-> 16 B

#define LDS_STRIDE 80       // 64 B of K-data + 16 B pad -> conflict-free ds_load_b128
#define GU_BUF     20480    // gateup per-stage LDS: A(128*80) + Bg(64*80) + Bu(64*80)
#define DN_BUF     20480    // down  per-stage LDS: A(128*80) + B(128*80)

__device__ __forceinline__ void atomic_add_f32(float* p, float v) {
    __hip_atomic_fetch_add(p, v, __ATOMIC_RELAXED, __HIP_MEMORY_SCOPE_AGENT);
}

__device__ __forceinline__ float fast_silu(float g) {
    // g * 1/(1+exp(-g)) via v_rcp_f32 (avoid IEEE div_scale/div_fixup sequence)
    return g * __builtin_amdgcn_rcpf(1.f + __expf(-g));
}

// ---- Tensor Data Mover: 2D tile (32 bf16 x 128 rows) global -> LDS, with LDS
// padding of 16 B after every 64 B so rows land on the 80-B padded stride.
// Rows >= rows_avail are zero-filled by TDM OOB handling (no clamp needed).
__device__ __forceinline__ void tdm_load_a_tile(unsigned lds_addr,
                                                const bf16_t* gptr,
                                                int rows_avail,
                                                int stride_elems) {
    unsigned long long ga = (unsigned long long)(uintptr_t)gptr;
    v4u g0;
    g0.x = 1u;                                   // count=1, user descriptor, no gather
    g0.y = lds_addr;                             // LDS byte address
    g0.z = (unsigned)(ga & 0xFFFFFFFFu);         // global_addr[31:0]
    g0.w = (unsigned)((ga >> 32) & 0x01FFFFFFu)  // global_addr[56:32]
         | (2u << 30);                           // type=2 (image)
    v8i g1;
    g1[0] = (int)((1u << 16)      // data_size = 2 bytes (bf16)
                | (1u << 20)      // pad_enable
                | (3u << 22)      // pad_interval: 16 DWORDs (64 B)
                | (3u << 25));    // pad_amount:   4 DWORDs (16 B)
    unsigned td0 = 32u;                               // tensor_dim0 (elements)
    unsigned td1 = (unsigned)rows_avail;              // tensor_dim1 (rows)
    g1[1] = (int)((td0 & 0xFFFFu) << 16);                                // dim0[15:0]
    g1[2] = (int)(((td0 >> 16) & 0xFFFFu) | ((td1 & 0xFFFFu) << 16));    // dim0[31:16], dim1[15:0]
    g1[3] = (int)(((td1 >> 16) & 0xFFFFu) | (32u << 16));                // dim1[31:16], tile_dim0=32
    g1[4] = 128;                                                         // tile_dim1=128, tile_dim2=0
    g1[5] = stride_elems;                                                // tensor_dim0_stride[31:0]
    g1[6] = 0;                                                           // stride[47:32], dim1_stride lo
    g1[7] = 0;
    v4i z4 = {0, 0, 0, 0};
    v8i z8 = {0, 0, 0, 0, 0, 0, 0, 0};
    // 6-arg form (upstream clang-23 / therock HIP headers)
    __builtin_amdgcn_tensor_load_to_lds(g0, g1, z4, z4, z8, 0);
}

// ---------------- init: zero output + expert counts ----------------
__global__ __launch_bounds__(256) void moe_zero_kernel(float* __restrict__ out,
                                                       int* __restrict__ counts) {
    long i = ((long)blockIdx.x * blockDim.x + threadIdx.x) * 4;  // 4 float4 / thread
    float4 z = {0.f, 0.f, 0.f, 0.f};
    float4* p = (float4*)out + i;
    p[0] = z; p[1] = z; p[2] = z; p[3] = z;
    if (blockIdx.x == 0 && threadIdx.x < E_NUM) counts[threadIdx.x] = 0;
}

// ---------------- cast activations fp32 -> bf16 ----------------
__global__ __launch_bounds__(256) void moe_cast_kernel(const float* __restrict__ x,
                                                       bf16_t* __restrict__ xb) {
    long i = ((long)blockIdx.x * blockDim.x + threadIdx.x) * 8;
    float4 a = *(const float4*)(x + i);
    float4 b = *(const float4*)(x + i + 4);
    Pack8 u;
    u.h[0] = (bf16_t)a.x; u.h[1] = (bf16_t)a.y; u.h[2] = (bf16_t)a.z; u.h[3] = (bf16_t)a.w;
    u.h[4] = (bf16_t)b.x; u.h[5] = (bf16_t)b.y; u.h[6] = (bf16_t)b.z; u.h[7] = (bf16_t)b.w;
    *(uint4*)(xb + i) = u.q;
}

// ---------------- router: one wave32 per token ----------------
__global__ __launch_bounds__(256)
void moe_router_kernel(const float* __restrict__ x,
                       const float* __restrict__ gate_w,      // [E,H]
                       const float* __restrict__ sh_gate_w,   // [1,H]
                       int*   __restrict__ counts,            // [E]
                       int*   __restrict__ tok_list,          // [E,T]
                       float* __restrict__ wt_list,           // [E,T]
                       float* __restrict__ sgate)             // [T]
{
    const int lane = threadIdx.x & 31;
    const int wid  = threadIdx.x >> 5;
    const int t    = blockIdx.x * 8 + wid;

    float acc[E_NUM];
    #pragma unroll
    for (int e = 0; e < E_NUM; ++e) acc[e] = 0.f;
    float accs = 0.f;

    for (int i = 0; i < H_DIM / 32; ++i) {
        const int h = i * 32 + lane;
        const float xv = x[(long)t * H_DIM + h];
        #pragma unroll
        for (int e = 0; e < E_NUM; ++e) acc[e] += xv * gate_w[e * H_DIM + h];
        accs += xv * sh_gate_w[h];
    }
    #pragma unroll
    for (int m = 16; m >= 1; m >>= 1) {
        #pragma unroll
        for (int e = 0; e < E_NUM; ++e) acc[e] += __shfl_xor(acc[e], m);
        accs += __shfl_xor(accs, m);
    }

    if (lane == 0) {
        float mx = acc[0];
        #pragma unroll
        for (int e = 1; e < E_NUM; ++e) mx = fmaxf(mx, acc[e]);
        float p[E_NUM]; float s = 0.f;
        #pragma unroll
        for (int e = 0; e < E_NUM; ++e) { p[e] = __expf(acc[e] - mx); s += p[e]; }
        const float inv = __builtin_amdgcn_rcpf(s);
        #pragma unroll
        for (int e = 0; e < E_NUM; ++e) p[e] *= inv;
        // top-4 (probs monotone in logits; renorm == softmax over selected)
        int   sel[TOPK]; float wv[TOPK]; float ssum = 0.f;
        #pragma unroll
        for (int k = 0; k < TOPK; ++k) {
            int be = 0; float bv = -1.f;
            #pragma unroll
            for (int e = 0; e < E_NUM; ++e) if (p[e] > bv) { bv = p[e]; be = e; }
            sel[k] = be; wv[k] = bv; ssum += bv; p[be] = -2.f;
        }
        const float rinv = 1.f / ssum;   // accuracy matters here; 4-term renorm
        #pragma unroll
        for (int k = 0; k < TOPK; ++k) {
            const int e = sel[k];
            const int pos = atomicAdd(&counts[e], 1);
            tok_list[e * T_TOK + pos] = t;
            wt_list [e * T_TOK + pos] = wv[k] * rinv;
        }
        sgate[t] = __builtin_amdgcn_rcpf(1.f + __expf(-accs));
    }
}

// ---------------- 16-wide exclusive prefix sum ----------------
__global__ void moe_offsets_kernel(const int* __restrict__ counts,
                                   int* __restrict__ offsets) {
    if (threadIdx.x == 0) {
        int s = 0;
        for (int e = 0; e < E_NUM; ++e) { offsets[e] = s; s += counts[e]; }
    }
}

// ---------------- fused gate+up+SiLU: Hout = silu(X Wg^T) * (X Wu^T) ----------------
// Block tile 128(M) x 64(N), K-step 32, 8 waves (4x2, each 32x32), double-buffered LDS.
__global__ __launch_bounds__(256)
void moe_gateup_kernel(const bf16_t* __restrict__ X,        // [T,H] bf16
                       const float*  __restrict__ Wg_base,  // [(E,)F,H] fp32
                       const float*  __restrict__ Wu_base,
                       long          w_estride,             // F*H per expert (0 shared)
                       bf16_t*       __restrict__ Hout,     // rows compacted per expert
                       int F,
                       const int* __restrict__ tok_list,    // null => identity gather
                       const int* __restrict__ counts,      // null => T
                       const int* __restrict__ offsets,     // null => 0
                       int mtiles)
{
    __shared__ __align__(16) unsigned char smem[2 * GU_BUF];

    const int e     = blockIdx.x / mtiles;
    const int m0    = (blockIdx.x % mtiles) * 128;
    const int n0    = blockIdx.y * 64;
    const int count = counts ? counts[e] : T_TOK;
    if (m0 >= count) return;                                   // block-uniform
    const int rowbase = offsets ? offsets[e] : 0;
    const float* Wg = Wg_base + (long)e * w_estride;
    const float* Wu = Wu_base + (long)e * w_estride;
    const int* tlist = tok_list ? (tok_list + e * T_TOK) : nullptr;

    const int tid  = threadIdx.x;
    const int lane = tid & 31;
    const int wid  = tid >> 5;
    const int waveM = wid & 3;
    const int waveN = wid >> 2;

    // A-tile map: 128 rows x 32 bf16; 32 B (half-row) per thread, gathered rows
    const int a_row  = tid >> 1;
    const int a_half = tid & 1;
    const int a_src  = (m0 + a_row < count) ? (m0 + a_row) : (count - 1);
    const int a_tok  = tlist ? tlist[a_src] : a_src;
    const bf16_t* a_gp = X + (long)a_tok * H_DIM + a_half * 16;

    // W-tile map: 64 rows x 32 fp32; 8 floats (32 B) per thread
    const int b_row = tid >> 2;
    const int b_seg = tid & 3;
    const float* wg_gp = Wg + (long)(n0 + b_row) * H_DIM + b_seg * 8;
    const float* wu_gp = Wu + (long)(n0 + b_row) * H_DIM + b_seg * 8;

    v8f accg[2][2], accu[2][2];
    #pragma unroll
    for (int i = 0; i < 2; ++i)
        #pragma unroll
        for (int j = 0; j < 2; ++j) {
            accg[i][j] = (v8f){0.f,0.f,0.f,0.f,0.f,0.f,0.f,0.f};
            accu[i][j] = (v8f){0.f,0.f,0.f,0.f,0.f,0.f,0.f,0.f};
        }

    auto load_stage = [&](int kt, int buf) {
        const int k0 = kt * 32;
        unsigned char* base  = smem + buf * GU_BUF;
        uint4 a0 = *(const uint4*)(a_gp + k0);
        uint4 a1 = *(const uint4*)(a_gp + k0 + 8);
        float4 g0 = *(const float4*)(wg_gp + k0);
        float4 g1 = *(const float4*)(wg_gp + k0 + 4);
        float4 u0 = *(const float4*)(wu_gp + k0);
        float4 u1 = *(const float4*)(wu_gp + k0 + 4);
        __builtin_prefetch(wg_gp + k0 + 64, 0, 1);   // 2 K-tiles ahead
        __builtin_prefetch(wu_gp + k0 + 64, 0, 1);
        Pack8 pg, pu;
        pg.h[0]=(bf16_t)g0.x; pg.h[1]=(bf16_t)g0.y; pg.h[2]=(bf16_t)g0.z; pg.h[3]=(bf16_t)g0.w;
        pg.h[4]=(bf16_t)g1.x; pg.h[5]=(bf16_t)g1.y; pg.h[6]=(bf16_t)g1.z; pg.h[7]=(bf16_t)g1.w;
        pu.h[0]=(bf16_t)u0.x; pu.h[1]=(bf16_t)u0.y; pu.h[2]=(bf16_t)u0.z; pu.h[3]=(bf16_t)u0.w;
        pu.h[4]=(bf16_t)u1.x; pu.h[5]=(bf16_t)u1.y; pu.h[6]=(bf16_t)u1.z; pu.h[7]=(bf16_t)u1.w;
        *(uint4*)(base + a_row * LDS_STRIDE + a_half * 32)      = a0;
        *(uint4*)(base + a_row * LDS_STRIDE + a_half * 32 + 16) = a1;
        *(uint4*)(base + 10240 + b_row * LDS_STRIDE + b_seg * 16) = pg.q;
        *(uint4*)(base + 15360 + b_row * LDS_STRIDE + b_seg * 16) = pu.q;
    };

    load_stage(0, 0);
    __syncthreads();

    const int KT = H_DIM / 32;
    for (int kt = 0; kt < KT; ++kt) {
        const int cur = kt & 1;
        if (kt + 1 < KT) load_stage(kt + 1, cur ^ 1);

        const unsigned char* base = smem + cur * GU_BUF;
        const int mrow = lane & 15;
        const int kh   = lane >> 4;
        FragAB af[2], bg[2], bu[2];
        #pragma unroll
        for (int mt = 0; mt < 2; ++mt) {
            const unsigned char* p = base + (waveM * 32 + mt * 16 + mrow) * LDS_STRIDE + kh * 32;
            af[mt].q[0] = *(const uint4*)p;
            af[mt].q[1] = *(const uint4*)(p + 16);
        }
        #pragma unroll
        for (int nt = 0; nt < 2; ++nt) {
            const unsigned char* pg2 = base + 10240 + (waveN * 32 + nt * 16 + mrow) * LDS_STRIDE + kh * 32;
            const unsigned char* pu2 = base + 15360 + (waveN * 32 + nt * 16 + mrow) * LDS_STRIDE + kh * 32;
            bg[nt].q[0] = *(const uint4*)pg2; bg[nt].q[1] = *(const uint4*)(pg2 + 16);
            bu[nt].q[0] = *(const uint4*)pu2; bu[nt].q[1] = *(const uint4*)(pu2 + 16);
        }
        #pragma unroll
        for (int mt = 0; mt < 2; ++mt)
            #pragma unroll
            for (int nt = 0; nt < 2; ++nt) {
                accg[mt][nt] = __builtin_amdgcn_wmma_f32_16x16x32_bf16(
                    false, af[mt].v, false, bg[nt].v, (short)0, accg[mt][nt], false, false);
                accu[mt][nt] = __builtin_amdgcn_wmma_f32_16x16x32_bf16(
                    false, af[mt].v, false, bu[nt].v, (short)0, accu[mt][nt], false, false);
            }
        __syncthreads();
    }

    // epilogue: silu(g)*u -> bf16 intermediate
    const int nloc = lane & 15;
    const int msel = (lane >> 4) * 8;
    #pragma unroll
    for (int mt = 0; mt < 2; ++mt)
        #pragma unroll
        for (int nt = 0; nt < 2; ++nt) {
            FragC cg, cu; cg.v = accg[mt][nt]; cu.v = accu[mt][nt];
            #pragma unroll
            for (int r = 0; r < 8; ++r) {
                const int grow = m0 + waveM * 32 + mt * 16 + msel + r;
                if (grow < count) {
                    const float h = fast_silu(cg.f[r]) * cu.f[r];
                    const int col = n0 + waveN * 32 + nt * 16 + nloc;
                    Hout[(long)(rowbase + grow) * F + col] = (bf16_t)h;
                }
            }
        }
}

// ---------------- down-projection: out[tok] += scale * (Hin row @ Wd^T) ----------------
// Block tile 128(M) x 128(N), K-step 32, 8 waves (4x2, each 32x64).
// A tile (contiguous bf16 rows) is DMA'd by the Tensor Data Mover with LDS padding;
// B tile (fp32 weights) is loaded/converted by the waves. Double-buffered.
__global__ __launch_bounds__(256)
void moe_down_kernel(const bf16_t* __restrict__ Hin,       // compacted rows, stride Kd
                     const float*  __restrict__ W_base,    // [(E,)H,Kd] fp32
                     long w_estride,
                     float* __restrict__ out,              // [T,H] fp32 (atomic acc)
                     int Kd,
                     const int* __restrict__ tok_list,
                     const int* __restrict__ counts,
                     const int* __restrict__ offsets,
                     const float* __restrict__ wt_list,    // expert scale, or null
                     const float* __restrict__ sgate,      // shared scale, or null
                     int mtiles)
{
    __shared__ __align__(16) unsigned char smem[2 * DN_BUF];

    const int e     = blockIdx.x / mtiles;
    const int m0    = (blockIdx.x % mtiles) * 128;
    const int n0    = blockIdx.y * 128;
    const int count = counts ? counts[e] : T_TOK;
    if (m0 >= count) return;
    const int rowbase = offsets ? offsets[e] : 0;
    const float* W = W_base + (long)e * w_estride;
    const int* tlist = tok_list ? (tok_list + e * T_TOK) : nullptr;
    const float* wts = wt_list ? (wt_list + e * T_TOK) : nullptr;

    const int tid  = threadIdx.x;
    const int lane = tid & 31;
    const int wid  = tid >> 5;
    const int waveM = wid & 3;
    const int waveN = wid >> 2;

    const bf16_t* A_base   = Hin + (long)(rowbase + m0) * Kd;
    const int     rows_avl = count - m0;     // TDM zero-fills rows beyond this

    // W-tile map: 128 rows x 32 fp32; 16 floats (64 B) per thread
    const int b_row = tid >> 1;
    const int b_seg = tid & 1;
    const float* w_gp = W + (long)(n0 + b_row) * Kd + b_seg * 16;

    v8f acc[2][4];
    #pragma unroll
    for (int i = 0; i < 2; ++i)
        #pragma unroll
        for (int j = 0; j < 4; ++j)
            acc[i][j] = (v8f){0.f,0.f,0.f,0.f,0.f,0.f,0.f,0.f};

    auto load_stage = [&](int kt, int buf) {
        unsigned char* base = smem + buf * DN_BUF;
        if (wid == 0)                         // one TDM issue per block per stage
            tdm_load_a_tile((unsigned)(uintptr_t)(void*)base,
                            A_base + kt * 32, rows_avl, Kd);
        const int k0 = kt * 32;
        float4 w0 = *(const float4*)(w_gp + k0);
        float4 w1 = *(const float4*)(w_gp + k0 + 4);
        float4 w2 = *(const float4*)(w_gp + k0 + 8);
        float4 w3 = *(const float4*)(w_gp + k0 + 12);
        __builtin_prefetch(w_gp + k0 + 64, 0, 1);
        Pack8 p0, p1;
        p0.h[0]=(bf16_t)w0.x; p0.h[1]=(bf16_t)w0.y; p0.h[2]=(bf16_t)w0.z; p0.h[3]=(bf16_t)w0.w;
        p0.h[4]=(bf16_t)w1.x; p0.h[5]=(bf16_t)w1.y; p0.h[6]=(bf16_t)w1.z; p0.h[7]=(bf16_t)w1.w;
        p1.h[0]=(bf16_t)w2.x; p1.h[1]=(bf16_t)w2.y; p1.h[2]=(bf16_t)w2.z; p1.h[3]=(bf16_t)w2.w;
        p1.h[4]=(bf16_t)w3.x; p1.h[5]=(bf16_t)w3.y; p1.h[6]=(bf16_t)w3.z; p1.h[7]=(bf16_t)w3.w;
        *(uint4*)(base + 10240 + b_row * LDS_STRIDE + b_seg * 32)      = p0.q;
        *(uint4*)(base + 10240 + b_row * LDS_STRIDE + b_seg * 32 + 16) = p1.q;
    };

    load_stage(0, 0);
    if (wid == 0) __builtin_amdgcn_s_wait_tensorcnt(0);
    __syncthreads();

    const int KT = Kd / 32;
    for (int kt = 0; kt < KT; ++kt) {
        const int cur = kt & 1;
        if (kt + 1 < KT) load_stage(kt + 1, cur ^ 1);

        const unsigned char* base = smem + cur * DN_BUF;
        const int mrow = lane & 15;
        const int kh   = lane >> 4;
        FragAB af[2], bf[4];
        #pragma unroll
        for (int mt = 0; mt < 2; ++mt) {
            const unsigned char* p = base + (waveM * 32 + mt * 16 + mrow) * LDS_STRIDE + kh * 32;
            af[mt].q[0] = *(const uint4*)p;
            af[mt].q[1] = *(const uint4*)(p + 16);
        }
        #pragma unroll
        for (int nt = 0; nt < 4; ++nt) {
            const unsigned char* p = base + 10240 + (waveN * 64 + nt * 16 + mrow) * LDS_STRIDE + kh * 32;
            bf[nt].q[0] = *(const uint4*)p;
            bf[nt].q[1] = *(const uint4*)(p + 16);
        }
        #pragma unroll
        for (int mt = 0; mt < 2; ++mt)
            #pragma unroll
            for (int nt = 0; nt < 4; ++nt)
                acc[mt][nt] = __builtin_amdgcn_wmma_f32_16x16x32_bf16(
                    false, af[mt].v, false, bf[nt].v, (short)0, acc[mt][nt], false, false);

        if (wid == 0 && kt + 1 < KT) __builtin_amdgcn_s_wait_tensorcnt(0);
        __syncthreads();
    }

    // epilogue: scale + atomic accumulate into out[tok]
    const int nloc = lane & 15;
    const int msel = (lane >> 4) * 8;
    #pragma unroll
    for (int mt = 0; mt < 2; ++mt) {
        FragC c[4];
        #pragma unroll
        for (int nt = 0; nt < 4; ++nt) c[nt].v = acc[mt][nt];
        #pragma unroll
        for (int r = 0; r < 8; ++r) {
            const int grow = m0 + waveM * 32 + mt * 16 + msel + r;
            if (grow < count) {
                const int   tok   = tlist ? tlist[grow] : grow;
                const float scale = wts ? wts[grow] : sgate[tok];
                #pragma unroll
                for (int nt = 0; nt < 4; ++nt) {
                    const int col = n0 + waveN * 64 + nt * 16 + nloc;
                    atomic_add_f32(&out[(long)tok * H_DIM + col], c[nt].f[r] * scale);
                }
            }
        }
    }
}

// ---------------- host-side launch ----------------
extern "C" void kernel_launch(void* const* d_in, const int* in_sizes, int n_in,
                              void* d_out, int out_size, void* d_ws, size_t ws_size,
                              hipStream_t stream) {
    const float* x       = (const float*)d_in[0];
    const float* gate_w  = (const float*)d_in[1];
    const float* e_gate  = (const float*)d_in[2];
    const float* e_up    = (const float*)d_in[3];
    const float* e_down  = (const float*)d_in[4];
    const float* s_gate  = (const float*)d_in[5];
    const float* s_up    = (const float*)d_in[6];
    const float* s_down  = (const float*)d_in[7];
    const float* s_egate = (const float*)d_in[8];
    float* out = (float*)d_out;

    char* ws = (char*)d_ws;
    bf16_t* x_bf   = (bf16_t*)ws;                                   // 16 MB
    bf16_t* hdn_e  = (bf16_t*)(ws + 16777216L);                     // 46.1 MB
    bf16_t* hdn_s  = (bf16_t*)(ws + 16777216L + 46137344L);         // 46.1 MB
    int*    counts = (int*)(ws + 109051904L);
    int*    offs   = counts + 16;
    int*    tok_l  = offs + 16;                                     // [E,T]
    float*  wt_l   = (float*)(tok_l + E_NUM * T_TOK);               // [E,T]
    float*  sgate  = wt_l + E_NUM * T_TOK;                          // [T]

    moe_zero_kernel  <<<2048, 256, 0, stream>>>(out, counts);
    moe_cast_kernel  <<<4096, 256, 0, stream>>>(x, x_bf);
    moe_router_kernel<<<T_TOK / 8, 256, 0, stream>>>(x, gate_w, s_egate,
                                                     counts, tok_l, wt_l, sgate);
    moe_offsets_kernel<<<1, 32, 0, stream>>>(counts, offs);

    // routed experts (top-4 sparsity; inactive token-tiles exit immediately)
    moe_gateup_kernel<<<dim3(E_NUM * (T_TOK / 128), FE_DIM / 64), 256, 0, stream>>>(
        x_bf, e_gate, e_up, (long)FE_DIM * H_DIM, hdn_e, FE_DIM,
        tok_l, counts, offs, T_TOK / 128);
    moe_down_kernel<<<dim3(E_NUM * (T_TOK / 128), H_DIM / 128), 256, 0, stream>>>(
        hdn_e, e_down, (long)H_DIM * FE_DIM, out, FE_DIM,
        tok_l, counts, offs, wt_l, nullptr, T_TOK / 128);

    // shared expert (dense, sigmoid-gated)
    moe_gateup_kernel<<<dim3(T_TOK / 128, FS_DIM / 64), 256, 0, stream>>>(
        x_bf, s_gate, s_up, 0L, hdn_s, FS_DIM,
        nullptr, nullptr, nullptr, T_TOK / 128);
    moe_down_kernel<<<dim3(T_TOK / 128, H_DIM / 128), 256, 0, stream>>>(
        hdn_s, s_down, 0L, out, FS_DIM,
        nullptr, nullptr, nullptr, nullptr, sgate, T_TOK / 128);
}
